// BidirectionalLSTM_65730179498211
// MI455X (gfx1250) — compile-verified, matrix operand order checked by reading, not run on hardware
//
#include <hip/hip_runtime.h>

// --- CDNA5 (gfx1250) wave32 WMMA types -------------------------------------
typedef __attribute__((ext_vector_type(16))) __bf16          v16bf;
typedef __attribute__((ext_vector_type(8)))  __bf16          v8bf;
typedef __attribute__((ext_vector_type(2)))  __bf16          v2bf;
typedef __attribute__((ext_vector_type(8)))  float           v8f;
typedef __attribute__((ext_vector_type(4)))  float           v4f;
typedef __attribute__((ext_vector_type(8)))  unsigned short  ush8;

constexpr int NB     = 32;    // N (batch)
constexpr int CIN    = 128;   // C
constexpr int TSTEPS = 2048;  // T
constexpr int HID    = 128;   // H
constexpr int G4     = 512;   // 4*H
// LDS row stride in bf16 elements: 136*2B = 272B = 68 dwords; 68 mod 64 = 4
// -> ds_load_b128 across 32 lanes touches each of the 64 banks exactly twice
// (conflict-free), while keeping rows 16B-aligned (272 = 17*16).
constexpr int LROW   = 136;

// Native fptrunc: lets the backend pick the hardware f32->bf16 convert.
__device__ __forceinline__ __bf16 f2bf(float f) {
  return static_cast<__bf16>(f);
}

// Two f32 -> packed 2 x bf16 in one dword (low = lo).
__device__ __forceinline__ unsigned pack_bf16x2(float lo, float hi) {
  v2bf p;
  p[0] = f2bf(lo);
  p[1] = f2bf(hi);
  return __builtin_bit_cast(unsigned, p);
}

__device__ __forceinline__ float fast_rcp(float v) {
#if __has_builtin(__builtin_amdgcn_rcpf)
  return __builtin_amdgcn_rcpf(v);              // v_rcp_f32
#else
  return 1.0f / v;
#endif
}

__device__ __forceinline__ float sigf(float v) {
  return fast_rcp(1.0f + __expf(-v));           // v_exp_f32 + v_rcp_f32
}

__device__ __forceinline__ float fast_tanh(float v) {
#if __has_builtin(__builtin_amdgcn_tanhf)
  return __builtin_amdgcn_tanhf(v);             // gfx1250 v_tanh_f32
#else
  return 1.0f - 2.0f * fast_rcp(__expf(2.0f * v) + 1.0f);
#endif
}

// Barrier that only fences LDS (DScnt): global stores drain in background.
__device__ __forceinline__ void lds_barrier() {
  asm volatile("s_wait_dscnt 0x0\n\t"
               "s_barrier_signal -1\n\t"
               "s_barrier_wait -1" ::: "memory");
}

// 16-bit B-matrix operand (32x16, col = lane%16) from a row-major bf16 LDS
// tile with padded row stride LROW; two ds_load_b128.
__device__ __forceinline__ v16bf loadB16(const unsigned short* __restrict__ src,
                                         int kb_elems, int l15) {
  ush8 r0 = *(const ush8*)(src + l15 * LROW + kb_elems);
  ush8 r1 = *(const ush8*)(src + l15 * LROW + kb_elems + 16);
  return __builtin_shufflevector(__builtin_bit_cast(v8bf, r0),
                                 __builtin_bit_cast(v8bf, r1),
                                 0, 1, 2, 3, 4, 5, 6, 7,
                                 8, 9, 10, 11, 12, 13, 14, 15);
}

// 16-bit A-matrix operand (16x32, row = lane%16) from a global fp32 W row,
// converted to bf16. Permuted gate row p -> original W row (p&3)*H + (p>>2).
__device__ __forceinline__ v16bf loadA_W(const float* __restrict__ Wrow,
                                         int kb, int loff) {
  v16bf a;
#pragma unroll
  for (int j = 0; j < 16; ++j) {
    const int k = kb + ((j < 8) ? j : (j + 8)) + loff;
    a[j] = f2bf(Wrow[k]);
  }
  return a;
}

// ============================================================================
// Kernel A: gxP[t][n][p] = sum_c x[n][c][t] * W_ih[g(p)][c] + b_ih + b_hh
// with p permuted: g(p) = (p&3)*H + (p>>2). Layout [T][N][4H] (p innermost)
// so the scan reads 8 consecutive floats per lane. Fully parallel WMMA GEMM.
// Grid: 4096 blocks (32 n x 128 t-tiles) x 256 threads (8 waves).
// ============================================================================
__global__ __launch_bounds__(256)
void lstm_gx_precompute(const float* __restrict__ x,      // [N, C, T]
                        const float* __restrict__ W_ih,   // [4H, C]
                        const float* __restrict__ b_ih,   // [4H]
                        const float* __restrict__ b_hh,   // [4H]
                        float* __restrict__ gx) {         // [T, N, 4H] permuted
  __shared__ alignas(16) unsigned short xstage[16 * LROW];  // bf16 [t][c]

  const int tid  = threadIdx.x;
  const int l    = tid & 31;
  const int l15  = l & 15;
  const int loff = (l & 16) ? 8 : 0;
  const int w    = tid >> 5;                 // wave 0..7

  const int n  = blockIdx.x >> 7;            // batch row
  const int tb = (blockIdx.x & 127) << 4;    // t-tile base

  // Stage x tile [16 t][128 c] as bf16 (coalesced along t).
#pragma unroll
  for (int ii = 0; ii < 8; ++ii) {
    const int idx = tid + ii * 256;          // 2048 elements
    const int t = idx & 15, c = idx >> 4;
    xstage[t * LROW + c] = __builtin_bit_cast(
        unsigned short,
        f2bf(x[(size_t)n * CIN * TSTEPS + (size_t)c * TSTEPS + tb + t]));
  }
  __syncthreads();

  // Each wave computes 4 gate tiles: p0 = 16*(w + 8*tt).
#pragma unroll
  for (int tt = 0; tt < 4; ++tt) {
    const int p0 = (w + tt * 8) << 4;
    const int pA = p0 + l15;
    const int gA = (pA & 3) * HID + (pA >> 2);
    const float* Wrow = W_ih + (size_t)gA * CIN;

    v8f acc;
#pragma unroll
    for (int v = 0; v < 8; ++v) {
      const int pv = p0 + loff + v;
      const int gv = (pv & 3) * HID + (pv >> 2);
      acc[v] = b_ih[gv] + b_hh[gv];
    }
#pragma unroll
    for (int kt = 0; kt < 4; ++kt) {
      v16bf a = loadA_W(Wrow, kt * 32, loff);
      v16bf b = loadB16(xstage, kt * 32 + loff, l15);
      acc = __builtin_amdgcn_wmma_f32_16x16x32_bf16(
          false, a, false, b, (short)0, acc, false, false);
    }
    // D: col = t (lane), row = p0 + loff + v -> 8 consecutive p values.
    const size_t base =
        ((size_t)(tb + l15) * NB + n) * G4 + p0 + loff;
    *(v4f*)&gx[base]     = __builtin_shufflevector(acc, acc, 0, 1, 2, 3);
    *(v4f*)&gx[base + 4] = __builtin_shufflevector(acc, acc, 4, 5, 6, 7);
  }
}

// ============================================================================
// Kernel B: recurrent scan consuming precomputed gx. Per step only the
// h * W_hh^T half remains: 4 WMMAs, 8 conflict-free ds_load_b128. The gx
// values are fetched with two global_load_b128 that overlap the whole WMMA
// phase (added during gate math, prefetched 2 steps ahead, L2-resident).
// One LDS-only barrier per step.
// ============================================================================
__global__ __launch_bounds__(1024)
void lstm_scan_gx(const float* __restrict__ gx,    // [T, N, 4H] permuted
                  const float* __restrict__ W_hh,  // [4H, H]
                  float* __restrict__ out) {       // [N, 2H, T]
  __shared__ alignas(16) unsigned short hbuf[2][16 * LROW];  // bf16 h tiles

  const int tid  = threadIdx.x;
  const int l    = tid & 31;
  const int l15  = l & 15;
  const int loff = (l & 16) ? 8 : 0;
  const int w    = tid >> 5;

  const int dir   = blockIdx.x >> 1;        // 0 = fwd, 1 = bwd (same weights)
  const int nbase = (blockIdx.x & 1) << 4;  // batch rows [nbase, nbase+16)

  // Pin permuted W_hh rows as WMMA A operands.
  const int pA = 16 * w + l15;
  const int gA = (pA & 3) * HID + (pA >> 2);
  v16bf Areg[4];
#pragma unroll
  for (int kt = 0; kt < 4; ++kt)
    Areg[kt] = loadA_W(W_hh + (size_t)gA * HID, kt * 32, loff);

  const int hh0 = (16 * w + loff) >> 2;  // even; lane owns hh0, hh0+1

  for (int idx = tid; idx < 16 * LROW; idx += 1024) hbuf[0][idx] = 0;
  lds_barrier();

  float c0 = 0.0f, c1 = 0.0f;  // cell state, register-resident

  for (int step = 0; step < TSTEPS; ++step) {
    const int r    = step & 1;
    const int tmem = dir ? (TSTEPS - 1 - step) : step;
    const unsigned short* hc = hbuf[r];

    // Issue gx loads first; they complete while the WMMAs run.
    const size_t gbase =
        ((size_t)tmem * NB + (nbase + l15)) * G4 + 16 * w + loff;
    const v4f ga = *(const v4f*)&gx[gbase];
    const v4f gb = *(const v4f*)&gx[gbase + 4];

    // 4 WMMAs (two independent 2-deep chains from inline-0 C), pipelined.
    v8f acc0, acc1;
#pragma unroll
    for (int v = 0; v < 8; ++v) { acc0[v] = 0.0f; acc1[v] = 0.0f; }

    v16bf bcur = loadB16(hc, loff, l15);
#pragma unroll
    for (int kt = 0; kt < 4; ++kt) {
      v16bf bnxt = bcur;
      if (kt < 3) bnxt = loadB16(hc, (kt + 1) * 32 + loff, l15);
      if (kt & 1) {
        acc1 = __builtin_amdgcn_wmma_f32_16x16x32_bf16(
            false, Areg[kt], false, bcur, (short)0, acc1, false, false);
      } else {
        acc0 = __builtin_amdgcn_wmma_f32_16x16x32_bf16(
            false, Areg[kt], false, bcur, (short)0, acc0, false, false);
      }
      bcur = bnxt;
    }

    // Gate math: recurrent part + precomputed input projection (gx).
    const float gi0 = acc0[0] + acc1[0] + ga[0], gf0 = acc0[1] + acc1[1] + ga[1];
    const float gg0 = acc0[2] + acc1[2] + ga[2], go0 = acc0[3] + acc1[3] + ga[3];
    const float gi1 = acc0[4] + acc1[4] + gb[0], gf1 = acc0[5] + acc1[5] + gb[1];
    const float gg1 = acc0[6] + acc1[6] + gb[2], go1 = acc0[7] + acc1[7] + gb[3];

    c0 = sigf(gf0) * c0 + sigf(gi0) * fast_tanh(gg0);
    c1 = sigf(gf1) * c1 + sigf(gi1) * fast_tanh(gg1);
    const float hn0 = sigf(go0) * fast_tanh(c0);
    const float hn1 = sigf(go1) * fast_tanh(c1);

    // h -> next LDS buffer (packed 2 x bf16; hh0 even; conflict-free banks).
    *(unsigned*)&hbuf[1 - r][l15 * LROW + hh0] = pack_bf16x2(hn0, hn1);

    // Output (background global stores; barrier does not wait on them).
    const size_t obase =
        ((size_t)(nbase + l15) * (2 * HID) + (size_t)dir * HID + hh0) * TSTEPS +
        tmem;
    out[obase] = hn0;
    out[obase + TSTEPS] = hn1;

    // Prefetch gx two steps ahead.
    const int tp = dir ? (tmem - 2) : (tmem + 2);
    if (tp >= 0 && tp < TSTEPS) {
      const size_t pbase =
          ((size_t)tp * NB + (nbase + l15)) * G4 + 16 * w + loff;
      __builtin_prefetch(&gx[pbase], 0, 1);  // global_prefetch_b8
    }
    lds_barrier();  // single barrier per timestep
  }
}

// ============================================================================
// Fallback: fully fused scan (no workspace): K=256 GEMM per step with x
// staged/double-buffered in LDS. Same register-resident gate scheme.
// ============================================================================
__global__ __launch_bounds__(1024)
void lstm_bidir_scan_fused(const float* __restrict__ x,
                           const float* __restrict__ W_ih,
                           const float* __restrict__ W_hh,
                           const float* __restrict__ b_ih,
                           const float* __restrict__ b_hh,
                           float* __restrict__ out) {
  __shared__ alignas(16) unsigned short xbuf[2][16 * LROW];
  __shared__ alignas(16) unsigned short hbuf[2][16 * LROW];

  const int tid  = threadIdx.x;
  const int l    = tid & 31;
  const int l15  = l & 15;
  const int loff = (l & 16) ? 8 : 0;
  const int w    = tid >> 5;

  const int dir   = blockIdx.x >> 1;
  const int nbase = (blockIdx.x & 1) << 4;

  const int pA = 16 * w + l15;
  const int gA = (pA & 3) * HID + (pA >> 2);
  v16bf Areg[8];
#pragma unroll
  for (int kt = 0; kt < 8; ++kt) {
    const float* Wrow = (kt < 4) ? (W_ih + (size_t)gA * CIN)
                                 : (W_hh + (size_t)gA * HID);
    Areg[kt] = loadA_W(Wrow, (kt & 3) * 32, loff);
  }

  v8f biasv;
#pragma unroll
  for (int v = 0; v < 8; ++v) {
    const int pv = 16 * w + loff + v;
    const int gv = (pv & 3) * HID + (pv >> 2);
    biasv[v] = b_ih[gv] + b_hh[gv];
  }

  const int hh0 = (16 * w + loff) >> 2;

  const int t0 = dir ? (TSTEPS - 1) : 0;
  for (int idx = tid; idx < 16 * LROW; idx += 1024) hbuf[0][idx] = 0;
#pragma unroll
  for (int ii = 0; ii < 2; ++ii) {
    const int idx = tid + ii * 1024;
    const int n = idx >> 7, cc = idx & 127;
    xbuf[0][n * LROW + cc] = __builtin_bit_cast(
        unsigned short,
        f2bf(x[(size_t)(nbase + n) * CIN * TSTEPS + (size_t)cc * TSTEPS + t0]));
  }
  lds_barrier();

  float c0 = 0.0f, c1 = 0.0f;

  for (int step = 0; step < TSTEPS; ++step) {
    const int r    = step & 1;
    const int tmem = dir ? (TSTEPS - 1 - step) : step;
    const unsigned short* xc = xbuf[r];
    const unsigned short* hc = hbuf[r];

    v8f acc0 = biasv, acc1;
#pragma unroll
    for (int v = 0; v < 8; ++v) acc1[v] = 0.0f;

    v16bf bcur = loadB16(xc, loff, l15);
#pragma unroll
    for (int kt = 0; kt < 8; ++kt) {
      v16bf bnxt = bcur;
      if (kt < 7) {
        const unsigned short* src = (kt + 1 < 4) ? xc : hc;
        bnxt = loadB16(src, ((kt + 1) & 3) * 32 + loff, l15);
      }
      if (kt & 1) {
        acc1 = __builtin_amdgcn_wmma_f32_16x16x32_bf16(
            false, Areg[kt], false, bcur, (short)0, acc1, false, false);
      } else {
        acc0 = __builtin_amdgcn_wmma_f32_16x16x32_bf16(
            false, Areg[kt], false, bcur, (short)0, acc0, false, false);
      }
      bcur = bnxt;
    }

    const float gi0 = acc0[0] + acc1[0], gf0 = acc0[1] + acc1[1];
    const float gg0 = acc0[2] + acc1[2], go0 = acc0[3] + acc1[3];
    const float gi1 = acc0[4] + acc1[4], gf1 = acc0[5] + acc1[5];
    const float gg1 = acc0[6] + acc1[6], go1 = acc0[7] + acc1[7];

    c0 = sigf(gf0) * c0 + sigf(gi0) * fast_tanh(gg0);
    c1 = sigf(gf1) * c1 + sigf(gi1) * fast_tanh(gg1);
    const float hn0 = sigf(go0) * fast_tanh(c0);
    const float hn1 = sigf(go1) * fast_tanh(c1);

    *(unsigned*)&hbuf[1 - r][l15 * LROW + hh0] = pack_bf16x2(hn0, hn1);

    const size_t obase =
        ((size_t)(nbase + l15) * (2 * HID) + (size_t)dir * HID + hh0) * TSTEPS +
        tmem;
    out[obase] = hn0;
    out[obase + TSTEPS] = hn1;

    if (step + 1 < TSTEPS) {
      const int tn = dir ? (tmem - 1) : (tmem + 1);
#pragma unroll
      for (int ii = 0; ii < 2; ++ii) {
        const int idx = tid + ii * 1024;
        const int n = idx >> 7, cc = idx & 127;
        const size_t xrow =
            (size_t)(nbase + n) * CIN * TSTEPS + (size_t)cc * TSTEPS;
        xbuf[1 - r][n * LROW + cc] =
            __builtin_bit_cast(unsigned short, f2bf(x[xrow + tn]));
        const int tn2 = dir ? (tn - 1) : (tn + 1);
        if (tn2 >= 0 && tn2 < TSTEPS)
          __builtin_prefetch(&x[xrow + tn2], 0, 1);
      }
    }
    lds_barrier();
  }
}

extern "C" void kernel_launch(void* const* d_in, const int* in_sizes, int n_in,
                              void* d_out, int out_size, void* d_ws, size_t ws_size,
                              hipStream_t stream) {
  (void)in_sizes; (void)n_in; (void)out_size;
  const float* x    = (const float*)d_in[0];
  const float* W_ih = (const float*)d_in[1];
  const float* W_hh = (const float*)d_in[2];
  const float* b_ih = (const float*)d_in[3];
  const float* b_hh = (const float*)d_in[4];
  float* out = (float*)d_out;

  const size_t gx_bytes = (size_t)TSTEPS * NB * G4 * sizeof(float);  // 128 MB
  if (d_ws != nullptr && ws_size >= gx_bytes) {
    float* gxb = (float*)d_ws;
    // Phase A: whole-sequence input projection (chip-wide WMMA GEMM).
    lstm_gx_precompute<<<dim3(NB * (TSTEPS / 16)), dim3(256), 0, stream>>>(
        x, W_ih, b_ih, b_hh, gxb);
    // Phase B: recurrent scan, 4 WMMAs/step.
    lstm_scan_gx<<<dim3(4), dim3(1024), 0, stream>>>(gxb, W_hh, out);
  } else {
    // Fallback: fused scan, 8 WMMAs/step, no workspace.
    lstm_bidir_scan_fused<<<dim3(4), dim3(1024), 0, stream>>>(
        x, W_ih, W_hh, b_ih, b_hh, out);
  }
}
